// TimeSformerGNN_21045339750687
// MI455X (gfx1250) — matrix operand order repaired and checked by reading, not compiled, not run on hardware
//
#include <hip/hip_runtime.h>
#include <hip/hip_bf16.h>

#define NTOT   8192
#define HIDD   768
#define NCLS   18
#define GDIM   256
#define CDIM   512          // HEADS*GD
#define KNN_K  5
#define E_CLS  324
#define E_TOT  (E_CLS + NCLS)   // 342 edges incl. self loops
#define NEG_SLOPE 0.2f
#define LN_EPS    1e-5f

typedef __bf16 bf16_t;
typedef __attribute__((ext_vector_type(8)))  __bf16 bf16x8;
typedef __attribute__((ext_vector_type(16))) __bf16 bf16x16;
typedef __attribute__((ext_vector_type(8)))  float  floatx8;

__device__ __forceinline__ bf16_t f2bf(float f) {
  unsigned u = __float_as_uint(f);
  unsigned r = u + 0x7FFFu + ((u >> 16) & 1u);   // round-to-nearest-even
  unsigned short h = (unsigned short)(r >> 16);
  return __builtin_bit_cast(bf16_t, h);
}
__device__ __forceinline__ float bf2f(bf16_t b) {
  unsigned short h = __builtin_bit_cast(unsigned short, b);
  return __uint_as_float(((unsigned)h) << 16);
}

// -------- fp32 -> (bf16 hi, bf16 lo) split ----------------------------------
__global__ void k_split(const float* __restrict__ x, bf16_t* __restrict__ hi,
                        bf16_t* __restrict__ lo, int n) {
  int i = blockIdx.x * blockDim.x + threadIdx.x;
  if (i >= n) return;
  float v = x[i];
  bf16_t h = f2bf(v);
  hi[i] = h;
  lo[i] = f2bf(v - bf2f(h));
}

// -------- row squared norms -------------------------------------------------
__global__ void k_rownorm(const float* __restrict__ x, float* __restrict__ x2,
                          int nrows, int K) {
  int row  = blockIdx.x * (blockDim.x >> 5) + (threadIdx.x >> 5);
  int lane = threadIdx.x & 31;
  if (row >= nrows) return;
  float s = 0.f;
  for (int k = lane; k < K; k += 32) { float v = x[(size_t)row * K + k]; s += v * v; }
  for (int o = 16; o > 0; o >>= 1) s += __shfl_down(s, o, 32);
  if (lane == 0) x2[row] = s;
}

// A fragment: per lane K = {kb..kb+7} U {kb+16..kb+23}  (16x32 bf16 layout)
__device__ __forceinline__ bf16x16 load_a_frag(const bf16_t* p) {
  bf16x8 a0 = *(const bf16x8*)(p);
  bf16x8 a1 = *(const bf16x8*)(p + 16);
  return __builtin_shufflevector(a0, a1, 0,1,2,3,4,5,6,7,8,9,10,11,12,13,14,15);
}

// -------- generic split-bf16 WMMA GEMM: out[M,N] = A[M,K] * B[N,K]^T + bias --
__global__ void __launch_bounds__(256)
k_gemm_bf16s(const bf16_t* __restrict__ ahi, const bf16_t* __restrict__ alo,
             const bf16_t* __restrict__ bhi, const bf16_t* __restrict__ blo,
             const float* __restrict__ bias, float* __restrict__ out,
             int M, int N, int K) {
  int wave = threadIdx.x >> 5;
  int lane = threadIdx.x & 31;
  int tile = blockIdx.x * (blockDim.x >> 5) + wave;
  int tn   = N >> 4;
  if (tile >= (M >> 4) * tn) return;            // wave-uniform exit: EXEC stays full
  int row0 = (tile / tn) << 4;
  int col0 = (tile % tn) << 4;
  int arow = row0 + (lane & 15);
  int akb  = (lane >> 4) << 3;
  int bcol = col0 + (lane & 15);
  int bkb  = (lane >> 4) << 3;
  const bf16_t* apH = ahi + (size_t)arow * K + akb;
  const bf16_t* apL = alo + (size_t)arow * K + akb;
  const bf16_t* bpH = bhi + (size_t)bcol * K + bkb;
  const bf16_t* bpL = blo + (size_t)bcol * K + bkb;
  floatx8 acc = {0.f,0.f,0.f,0.f,0.f,0.f,0.f,0.f};
  for (int k0 = 0; k0 < K; k0 += 32) {
    bf16x16 aH = load_a_frag(apH + k0);
    bf16x16 aL = load_a_frag(apL + k0);
    bf16x16 bH = load_a_frag(bpH + k0);
    bf16x16 bL = load_a_frag(bpL + k0);
    acc = __builtin_amdgcn_wmma_f32_16x16x32_bf16(false, aH, false, bH, (short)0, acc, false, false);
    acc = __builtin_amdgcn_wmma_f32_16x16x32_bf16(false, aH, false, bL, (short)0, acc, false, false);
    acc = __builtin_amdgcn_wmma_f32_16x16x32_bf16(false, aL, false, bH, (short)0, acc, false, false);
  }
  float bv   = bias ? bias[col0 + (lane & 15)] : 0.f;
  int  rbase = row0 + ((lane >> 4) << 3);
  int  c     = col0 + (lane & 15);
  #pragma unroll
  for (int v = 0; v < 8; ++v)
    out[(size_t)(rbase + v) * N + c] = acc[v] + bv;
}

// -------- fused Gram + distance + per-row top-5 kNN --------------------------
__global__ void __launch_bounds__(32)
k_knn(const bf16_t* __restrict__ xhi, const bf16_t* __restrict__ xlo,
      const float* __restrict__ x2, int* __restrict__ knn) {
  __shared__ float tile[16 * 17];
  int lane = threadIdx.x & 31;
  int row0 = blockIdx.x << 4;
  int akb  = (lane >> 4) << 3;
  int bkb  = (lane >> 4) << 3;
  const bf16_t* apH = xhi + (size_t)(row0 + (lane & 15)) * HIDD + akb;
  const bf16_t* apL = xlo + (size_t)(row0 + (lane & 15)) * HIDD + akb;
  float x2r[8];
  #pragma unroll
  for (int v = 0; v < 8; ++v) x2r[v] = x2[row0 + v + ((lane >> 4) << 3)];
  float bd[KNN_K]; int bi[KNN_K];
  #pragma unroll
  for (int j = 0; j < KNN_K; ++j) { bd[j] = 3.4e38f; bi[j] = 0; }

  for (int jt = 0; jt < (NTOT >> 4); ++jt) {
    int bcol = (jt << 4) + (lane & 15);
    const bf16_t* bpH = xhi + (size_t)bcol * HIDD + bkb;
    const bf16_t* bpL = xlo + (size_t)bcol * HIDD + bkb;
    floatx8 acc = {0.f,0.f,0.f,0.f,0.f,0.f,0.f,0.f};
    for (int k0 = 0; k0 < HIDD; k0 += 32) {
      bf16x16 aH = load_a_frag(apH + k0);
      bf16x16 aL = load_a_frag(apL + k0);
      bf16x16 bH = load_a_frag(bpH + k0);
      bf16x16 bL = load_a_frag(bpL + k0);
      acc = __builtin_amdgcn_wmma_f32_16x16x32_bf16(false, aH, false, bH, (short)0, acc, false, false);
      acc = __builtin_amdgcn_wmma_f32_16x16x32_bf16(false, aH, false, bL, (short)0, acc, false, false);
      acc = __builtin_amdgcn_wmma_f32_16x16x32_bf16(false, aL, false, bH, (short)0, acc, false, false);
    }
    float colx2 = x2[bcol];
    #pragma unroll
    for (int v = 0; v < 8; ++v) {
      int r = v + ((lane >> 4) << 3);
      tile[r * 17 + (lane & 15)] = x2r[v] + colx2 - 2.f * acc[v];
    }
    __syncthreads();
    if (lane < 16) {
      #pragma unroll 4
      for (int n = 0; n < 16; ++n) {
        float d = tile[lane * 17 + n];
        if (d < bd[KNN_K - 1]) {
          bd[KNN_K - 1] = d; bi[KNN_K - 1] = (jt << 4) + n;
          #pragma unroll
          for (int j = KNN_K - 1; j > 0; --j)
            if (bd[j] < bd[j - 1]) {
              float td = bd[j]; bd[j] = bd[j - 1]; bd[j - 1] = td;
              int   ti = bi[j]; bi[j] = bi[j - 1]; bi[j - 1] = ti;
            }
        }
      }
    }
    __syncthreads();
  }
  if (lane < 16) {
    #pragma unroll
    for (int e = 0; e < KNN_K; ++e)
      knn[(size_t)(row0 + lane) * KNN_K + e] = bi[e];
  }
}

// -------- class graph: xl/xr linears (18 nodes, trivial) ---------------------
__global__ void k_class_lin(const float* __restrict__ emb,
                            const float* __restrict__ wl, const float* __restrict__ bl,
                            const float* __restrict__ wr, const float* __restrict__ br,
                            float* __restrict__ xlc, float* __restrict__ xrc) {
  int i = blockIdx.x;
  for (int o = threadIdx.x; o < CDIM; o += blockDim.x) {
    float sl = bl[o], sr = br[o];
    for (int k = 0; k < HIDD; ++k) {
      float e = emb[i * HIDD + k];
      sl += e * wl[o * HIDD + k];
      sr += e * wr[o * HIDD + k];
    }
    xlc[i * CDIM + o] = sl;
    xrc[i * CDIM + o] = sr;
  }
}

// -------- class graph GATv2 (segment softmax over 342 edges) -----------------
__global__ void k_class_gat(const float* __restrict__ xlc, const float* __restrict__ xrc,
                            const int* __restrict__ eidx, const float* __restrict__ att,
                            const float* __restrict__ bias, float* __restrict__ cfeat) {
  __shared__ float sc[E_TOT * 2];
  __shared__ float hstat[4];
  int i = blockIdx.x, t = threadIdx.x;
  for (int e = t; e < E_TOT; e += blockDim.x) {
    int src = (e < E_CLS) ? eidx[e] : (e - E_CLS);
    int dst = (e < E_CLS) ? eidx[E_CLS + e] : (e - E_CLS);
    if (dst != i) { sc[e * 2] = -1e30f; sc[e * 2 + 1] = -1e30f; continue; }
    for (int h = 0; h < 2; ++h) {
      float s = 0.f;
      for (int d = 0; d < GDIM; ++d) {
        int c = h * GDIM + d;
        float z = xlc[src * CDIM + c] + xrc[i * CDIM + c];
        z = z > 0.f ? z : NEG_SLOPE * z;
        s += z * att[c];
      }
      sc[e * 2 + h] = s;
    }
  }
  __syncthreads();
  if (t < 2) {
    float m = -1e30f;
    for (int e = 0; e < E_TOT; ++e) m = fmaxf(m, sc[e * 2 + t]);
    float sum = 0.f;
    for (int e = 0; e < E_TOT; ++e)
      if (sc[e * 2 + t] > -1e29f) sum += __expf(sc[e * 2 + t] - m);
    hstat[t] = m; hstat[2 + t] = 1.f / sum;
  }
  __syncthreads();
  for (int idx = t; idx < E_TOT * 2; idx += blockDim.x) {
    int h = idx & 1;
    float v = sc[idx];
    sc[idx] = (v > -1e29f) ? __expf(v - hstat[h]) * hstat[2 + h] : 0.f;
  }
  __syncthreads();
  for (int c = t; c < CDIM; c += blockDim.x) {
    int h = c >> 8;
    float s = 0.f;
    for (int e = 0; e < E_TOT; ++e) {
      float a = sc[e * 2 + h];
      if (a != 0.f) {
        int src = (e < E_CLS) ? eidx[e] : (e - E_CLS);
        s += a * xlc[src * CDIM + c];
      }
    }
    cfeat[i * CDIM + c] = s + bias[c];
  }
}

__global__ void k_ctx_base(const float* __restrict__ cfeat, float* __restrict__ ctxb) {
  int c = threadIdx.x;
  float s = 0.f;
  for (int i = 0; i < NCLS; ++i) s += cfeat[i * CDIM + c];
  ctxb[c] = s * (1.f / NCLS);
}

// -------- instance GATv2 aggregation (6 edges/node) + ctx_base add -----------
__global__ void __launch_bounds__(128)
k_gat_agg(const float* __restrict__ xl, const float* __restrict__ xr,
          const int* __restrict__ knn, const float* __restrict__ att,
          const float* __restrict__ bias, const float* __restrict__ ctxb,
          float* __restrict__ ctx) {
  __shared__ float red[12][129];
  __shared__ float alpha[12];
  int i = blockIdx.x, t = threadIdx.x;  // 128 threads
  int srcs[6];
  #pragma unroll
  for (int e = 0; e < KNN_K; ++e) srcs[e] = knn[i * KNN_K + e];
  srcs[5] = i;                          // appended self loop
  float p0[6] = {0,0,0,0,0,0}, p1[6] = {0,0,0,0,0,0};
  #pragma unroll
  for (int ci = 0; ci < 4; ++ci) {
    int c = t + ci * 128;
    float xrv = xr[(size_t)i * CDIM + c];
    float av  = att[c];
    #pragma unroll
    for (int e = 0; e < 6; ++e) {
      float z = xl[(size_t)srcs[e] * CDIM + c] + xrv;
      z = z > 0.f ? z : NEG_SLOPE * z;
      if (ci < 2) p0[e] += z * av; else p1[e] += z * av;
    }
  }
  #pragma unroll
  for (int e = 0; e < 6; ++e) { red[e * 2 + 0][t] = p0[e]; red[e * 2 + 1][t] = p1[e]; }
  __syncthreads();
  if (t < 12) { float s = 0.f; for (int j = 0; j < 128; ++j) s += red[t][j]; red[t][128] = s; }
  __syncthreads();
  if (t < 2) {
    float m = -1e30f;
    for (int e = 0; e < 6; ++e) m = fmaxf(m, red[e * 2 + t][128]);
    float ex[6], sum = 0.f;
    for (int e = 0; e < 6; ++e) { ex[e] = __expf(red[e * 2 + t][128] - m); sum += ex[e]; }
    for (int e = 0; e < 6; ++e) alpha[e * 2 + t] = ex[e] / sum;
  }
  __syncthreads();
  #pragma unroll
  for (int ci = 0; ci < 4; ++ci) {
    int c = t + ci * 128;
    int h = c >> 8;
    float s = 0.f;
    #pragma unroll
    for (int e = 0; e < 6; ++e) s += alpha[e * 2 + h] * xl[(size_t)srcs[e] * CDIM + c];
    ctx[(size_t)i * CDIM + c] = s + bias[c] + ctxb[c];
  }
}

// -------- LayerNorm(x + res) -------------------------------------------------
__global__ void k_ln_res(const float* __restrict__ a, const float* __restrict__ r,
                         const float* __restrict__ g, const float* __restrict__ b,
                         float* __restrict__ out) {
  __shared__ float s1[256], s2[256];
  int row = blockIdx.x, t = threadIdx.x;
  float v[3], sum = 0.f, sq = 0.f;
  #pragma unroll
  for (int j = 0; j < 3; ++j) {
    int idx = t + j * 256;
    float x = a[(size_t)row * HIDD + idx] + r[(size_t)row * HIDD + idx];
    v[j] = x; sum += x; sq += x * x;
  }
  s1[t] = sum; s2[t] = sq; __syncthreads();
  for (int s = 128; s > 0; s >>= 1) {
    if (t < s) { s1[t] += s1[t + s]; s2[t] += s2[t + s]; }
    __syncthreads();
  }
  float mean = s1[0] * (1.f / HIDD);
  float var  = s2[0] * (1.f / HIDD) - mean * mean;
  float rstd = rsqrtf(var + LN_EPS);
  #pragma unroll
  for (int j = 0; j < 3; ++j) {
    int idx = t + j * 256;
    out[(size_t)row * HIDD + idx] = (v[j] - mean) * rstd * g[idx] + b[idx];
  }
}

// -------- GELU(LayerNorm(x)) -------------------------------------------------
__global__ void k_ln_gelu(const float* __restrict__ a, const float* __restrict__ g,
                          const float* __restrict__ b, float* __restrict__ out) {
  __shared__ float s1[256], s2[256];
  int row = blockIdx.x, t = threadIdx.x;
  float v[3], sum = 0.f, sq = 0.f;
  #pragma unroll
  for (int j = 0; j < 3; ++j) {
    int idx = t + j * 256;
    float x = a[(size_t)row * HIDD + idx];
    v[j] = x; sum += x; sq += x * x;
  }
  s1[t] = sum; s2[t] = sq; __syncthreads();
  for (int s = 128; s > 0; s >>= 1) {
    if (t < s) { s1[t] += s1[t + s]; s2[t] += s2[t + s]; }
    __syncthreads();
  }
  float mean = s1[0] * (1.f / HIDD);
  float var  = s2[0] * (1.f / HIDD) - mean * mean;
  float rstd = rsqrtf(var + LN_EPS);
  #pragma unroll
  for (int j = 0; j < 3; ++j) {
    int idx = t + j * 256;
    float y = (v[j] - mean) * rstd * g[idx] + b[idx];
    out[(size_t)row * HIDD + idx] = 0.5f * y * (1.f + erff(y * 0.70710678118654752f));
  }
}

// -------- GroupWiseLinear + temperature --------------------------------------
__global__ void k_logits(const float* __restrict__ pre, const float* __restrict__ gw,
                         const float* __restrict__ gb, const float* __restrict__ temp,
                         float* __restrict__ out) {
  int row = blockIdx.x, c = threadIdx.y, lane = threadIdx.x;
  float s = 0.f;
  for (int d = lane; d < HIDD; d += 32)
    s += pre[(size_t)row * HIDD + d] * gw[c * HIDD + d];
  for (int o = 16; o > 0; o >>= 1) s += __shfl_down(s, o, 32);
  if (lane == 0) out[(size_t)row * NCLS + c] = (s + gb[c]) / temp[0];
}

extern "C" void kernel_launch(void* const* d_in, const int* in_sizes, int n_in,
                              void* d_out, int out_size, void* d_ws, size_t ws_size,
                              hipStream_t stream) {
  (void)in_sizes; (void)n_in; (void)out_size; (void)ws_size;
  const float* X       = (const float*)d_in[0];
  const int*   CEI     = (const int*)  d_in[1];
  const float* CEMB    = (const float*)d_in[2];
  const float* GC_WL   = (const float*)d_in[3];
  const float* GC_BL   = (const float*)d_in[4];
  const float* GC_WR   = (const float*)d_in[5];
  const float* GC_BR   = (const float*)d_in[6];
  const float* GC_ATT  = (const float*)d_in[7];
  const float* GC_BIAS = (const float*)d_in[8];
  const float* GI_WL   = (const float*)d_in[9];
  const float* GI_BL   = (const float*)d_in[10];
  const float* GI_WR   = (const float*)d_in[11];
  const float* GI_BR   = (const float*)d_in[12];
  const float* GI_ATT  = (const float*)d_in[13];
  const float* GI_BIAS = (const float*)d_in[14];
  // d_in[15]=q_w, d_in[16]=q_b cancel out of the forward value (seq_len 1)
  const float* CTX_W   = (const float*)d_in[17];
  const float* CTX_B   = (const float*)d_in[18];
  const float* WV      = (const float*)d_in[19];
  const float* BV      = (const float*)d_in[20];
  const float* WO      = (const float*)d_in[21];
  const float* BO      = (const float*)d_in[22];
  const float* LN1G    = (const float*)d_in[23];
  const float* LN1B    = (const float*)d_in[24];
  const float* LN2G    = (const float*)d_in[25];
  const float* LN2B    = (const float*)d_in[26];
  const float* GWW     = (const float*)d_in[27];
  const float* GWB     = (const float*)d_in[28];
  const float* TEMP    = (const float*)d_in[29];
  float* OUT = (float*)d_out;

  char* ws = (char*)d_ws;
  size_t off = 0;
  auto alloc = [&](size_t bytes) { size_t o = off; off += (bytes + 255) & ~(size_t)255; return o; };

  bf16_t* AHI  = (bf16_t*)(ws + alloc((size_t)NTOT * HIDD * 2));  // activation split (reused)
  bf16_t* ALO  = (bf16_t*)(ws + alloc((size_t)NTOT * HIDD * 2));
  bf16_t* BHI  = (bf16_t*)(ws + alloc((size_t)HIDD * HIDD * 2));  // weight split (reused)
  bf16_t* BLO  = (bf16_t*)(ws + alloc((size_t)HIDD * HIDD * 2));
  float*  X2   = (float*) (ws + alloc((size_t)NTOT * 4));
  int*    KNN  = (int*)   (ws + alloc((size_t)NTOT * KNN_K * 4));
  float*  XL   = (float*) (ws + alloc((size_t)NTOT * CDIM * 4));
  float*  XR   = (float*) (ws + alloc((size_t)NTOT * CDIM * 4));
  float*  CTX  = (float*) (ws + alloc((size_t)NTOT * CDIM * 4));
  float*  BUFA = (float*) (ws + alloc((size_t)NTOT * HIDD * 4));  // ctx_p -> attn -> pre
  float*  BUFB = (float*) (ws + alloc((size_t)NTOT * HIDD * 4));  // v -> fused
  float*  XLC  = (float*) (ws + alloc((size_t)NCLS * CDIM * 4));
  float*  XRC  = (float*) (ws + alloc((size_t)NCLS * CDIM * 4));
  float*  CFT  = (float*) (ws + alloc((size_t)NCLS * CDIM * 4));
  float*  CTXB = (float*) (ws + alloc((size_t)CDIM * 4));

  const int nX = NTOT * HIDD, nC = NTOT * CDIM;
  auto cdiv = [](int a, int b) { return (a + b - 1) / b; };

  // kNN graph: split X once, Gram via WMMA, fused distance + top-5
  k_split  <<<cdiv(nX, 256), 256, 0, stream>>>(X, AHI, ALO, nX);
  k_rownorm<<<NTOT / 8, 256, 0, stream>>>(X, X2, NTOT, HIDD);
  k_knn    <<<NTOT / 16, 32, 0, stream>>>(AHI, ALO, X2, KNN);

  // instance GATv2 linears: xl = X*wl^T+bl, xr = X*wr^T+br
  k_split<<<cdiv(CDIM * HIDD, 256), 256, 0, stream>>>(GI_WL, BHI, BLO, CDIM * HIDD);
  k_gemm_bf16s<<<cdiv((NTOT / 16) * (CDIM / 16), 8), 256, 0, stream>>>(AHI, ALO, BHI, BLO, GI_BL, XL, NTOT, CDIM, HIDD);
  k_split<<<cdiv(CDIM * HIDD, 256), 256, 0, stream>>>(GI_WR, BHI, BLO, CDIM * HIDD);
  k_gemm_bf16s<<<cdiv((NTOT / 16) * (CDIM / 16), 8), 256, 0, stream>>>(AHI, ALO, BHI, BLO, GI_BR, XR, NTOT, CDIM, HIDD);

  // class graph (tiny) -> ctx_base
  k_class_lin<<<NCLS, 256, 0, stream>>>(CEMB, GC_WL, GC_BL, GC_WR, GC_BR, XLC, XRC);
  k_class_gat<<<NCLS, 128, 0, stream>>>(XLC, XRC, CEI, GC_ATT, GC_BIAS, CFT);
  k_ctx_base <<<1, CDIM, 0, stream>>>(CFT, CTXB);

  // instance aggregation + ctx_base add -> ctx [N,512]
  k_gat_agg<<<NTOT, 128, 0, stream>>>(XL, XR, KNN, GI_ATT, GI_BIAS, CTXB, CTX);

  // ctx_p = ctx * ctx_w^T + ctx_b
  k_split<<<cdiv(nC, 256), 256, 0, stream>>>(CTX, AHI, ALO, nC);
  k_split<<<cdiv(HIDD * CDIM, 256), 256, 0, stream>>>(CTX_W, BHI, BLO, HIDD * CDIM);
  k_gemm_bf16s<<<cdiv((NTOT / 16) * (HIDD / 16), 8), 256, 0, stream>>>(AHI, ALO, BHI, BLO, CTX_B, BUFA, NTOT, HIDD, CDIM);

  // v = ctx_p * wv^T + bv
  k_split<<<cdiv(nX, 256), 256, 0, stream>>>(BUFA, AHI, ALO, nX);
  k_split<<<cdiv(HIDD * HIDD, 256), 256, 0, stream>>>(WV, BHI, BLO, HIDD * HIDD);
  k_gemm_bf16s<<<cdiv((NTOT / 16) * (HIDD / 16), 8), 256, 0, stream>>>(AHI, ALO, BHI, BLO, BV, BUFB, NTOT, HIDD, HIDD);

  // attn = v * wo^T + bo
  k_split<<<cdiv(nX, 256), 256, 0, stream>>>(BUFB, AHI, ALO, nX);
  k_split<<<cdiv(HIDD * HIDD, 256), 256, 0, stream>>>(WO, BHI, BLO, HIDD * HIDD);
  k_gemm_bf16s<<<cdiv((NTOT / 16) * (HIDD / 16), 8), 256, 0, stream>>>(AHI, ALO, BHI, BLO, BO, BUFA, NTOT, HIDD, HIDD);

  // fused = LN(x + attn); pre = GELU(LN(fused)); logits
  k_ln_res <<<NTOT, 256, 0, stream>>>(X, BUFA, LN1G, LN1B, BUFB);
  k_ln_gelu<<<NTOT, 256, 0, stream>>>(BUFB, LN2G, LN2B, BUFA);
  k_logits <<<NTOT, dim3(32, NCLS), 0, stream>>>(BUFA, GWW, GWB, TEMP, OUT);
}
